// RNN_8839042695658
// MI455X (gfx1250) — matrix operand order repaired and checked by reading, not compiled, not run on hardware
//
#include <hip/hip_runtime.h>
#include <math.h>

// ---------------------------------------------------------------------------
// LSTM captioning loss for MI455X (gfx1250, wave32, WMMA).
// All GEMMs use v_wmma_f32_16x16x32_bf16 (fp32 accumulate). Weights are
// converted fp32->bf16 AND transposed to K-major once, so every WMMA operand
// fragment is assembled from contiguous b128 loads (no per-element packing).
// The vocab GEMM (4096x512x32000, ~134 GFLOP, dominant) is fused with an
// online logsumexp; the 524MB logits tensor is never materialized. W_vocab
// bf16 (32MB) stays resident in the 192MB L2 across row-tile re-reads.
// ---------------------------------------------------------------------------

typedef __bf16 bf16;
typedef __attribute__((ext_vector_type(16))) __bf16 v16bf;
typedef __attribute__((ext_vector_type(8)))  __bf16 v8bf;
typedef __attribute__((ext_vector_type(8)))  float  v8f;
typedef __attribute__((ext_vector_type(8)))  float  v8ff;

#define T_STEPS 256
#define NBATCH  16
#define TP1     257
#define DWVEC   256     // word-vector dim
#define HID     512
#define GATES   2048    // 4*HID
#define VOCABSZ 32000
#define NROWS   4096    // T_STEPS * NBATCH, row r = t*16 + n
#define VCHUNK  256     // vocab columns per workgroup
#define NCHUNKS 125     // VOCABSZ / VCHUNK

// --- WMMA fragment helpers (A/B share one contiguous k order; C/D as in ISA) -------
static __device__ __forceinline__ v16bf cat8(v8bf lo, v8bf hi) {
  return __builtin_shufflevector(lo, hi, 0, 1, 2, 3, 4, 5, 6, 7,
                                 8, 9, 10, 11, 12, 13, 14, 15);
}

// A 16x32 bf16, row-major source [m][K]: lane m = lane&15;
// elems e -> K = 32kk + (lane&16?16:0) + e (0..15).
static __device__ __forceinline__ v16bf loadA_bf(const bf16* row, int kk, int lane) {
  const bf16* p = row + (kk << 5) + ((lane & 16) ? 16 : 0);
  v8bf lo = *(const v8bf*)p;          // 16B contiguous
  v8bf hi = *(const v8bf*)(p + 8);    // 16B contiguous
  return cat8(lo, hi);
}

// Same A fragment gathered from an fp32 row (embedding), converted on load.
static __device__ __forceinline__ v16bf loadA_f32(const float* row, int kk, int lane) {
  const float* p = row + (kk << 5) + ((lane & 16) ? 16 : 0);
  v8ff lo = *(const v8ff*)p;          // 32B contiguous
  v8ff hi = *(const v8ff*)(p + 8);
  return cat8(__builtin_convertvector(lo, v8bf),
              __builtin_convertvector(hi, v8bf));
}

// B 32x16 bf16 from K-major (transposed) weights Wt[n][K]: lane n = lane&15;
// elems e -> K = 32kk + (lane&16?16:0) + e  => 16 consecutive bf16 = 32B.
static __device__ __forceinline__ v16bf loadB_bf(const bf16* Wt, long n, int K,
                                                 int kk, int lane) {
  const bf16* p = Wt + n * (long)K + (kk << 5) + ((lane & 16) ? 16 : 0);
  return *(const v16bf*)p;            // 32B aligned for all K used here
}

// C/D f32: VGPR p -> row p (lanes 0-15) or p+8 (lanes 16-31); col = lane&15
static __device__ __forceinline__ int c_row(int lane, int p) {
  return p + ((lane & 16) ? 8 : 0);
}

static __device__ __forceinline__ v8f wmma_bf16(v16bf a, v16bf b, v8f c) {
  return __builtin_amdgcn_wmma_f32_16x16x32_bf16(false, a, false, b,
                                                 (short)0, c, false, false);
}

// --- fp32(KxN) -> bf16(NxK) convert+transpose, LDS tiled -------------------
// grid (K/32, N/32), block 256 (32x8)
__global__ void cvt_transpose_bf16(const float* __restrict__ in,
                                   bf16* __restrict__ out, int K, int N) {
  __shared__ float tile[32][33];
  const int kt = blockIdx.x * 32, nt = blockIdx.y * 32;
  const int tx = threadIdx.x & 31, ty = threadIdx.x >> 5;
  for (int i = ty; i < 32; i += 8)
    tile[i][tx] = in[(long)(kt + i) * N + nt + tx];
  __syncthreads();
  for (int i = ty; i < 32; i += 8)
    out[(long)(nt + i) * K + kt + tx] = (bf16)tile[tx][i];
}

// --- Kernel 1: xw[r, :] = W_embed[token(r)] @ Wx + b  (r = t*16+n) ---------
// grid (GATES/256=8, 256 row tiles), block 256 (8 waves, 2 col tiles/wave)
__global__ void embed_xw(const int* __restrict__ captions,
                         const float* __restrict__ W_embed,
                         const bf16* __restrict__ Wx_t,   // [GATES][DWVEC]
                         const float* __restrict__ bgate,
                         float* __restrict__ xw) {
  const int lane = threadIdx.x & 31;
  const int wave = threadIdx.x >> 5;
  const int rt = blockIdx.y;                       // row tile == timestep t
  const int m = lane & 15;                         // row within tile == batch n
  const int token = captions[m * TP1 + rt];
  const float* erow = W_embed + (long)token * DWVEC;
  const int n0 = blockIdx.x * 256 + wave * 32 + (lane & 15);

  v8f acc0 = {}, acc1 = {};
  for (int kk = 0; kk < DWVEC / 32; ++kk) {
    v16bf a = loadA_f32(erow, kk, lane);
    v16bf b0 = loadB_bf(Wx_t, n0, DWVEC, kk, lane);
    v16bf b1 = loadB_bf(Wx_t, n0 + 16, DWVEC, kk, lane);
    acc0 = wmma_bf16(a, b0, acc0);
    acc1 = wmma_bf16(a, b1, acc1);
  }
#pragma unroll
  for (int p = 0; p < 8; ++p) {
    int r = rt * 16 + c_row(lane, p);
    xw[(long)r * GATES + n0]      = acc0[p] + bgate[n0];
    xw[(long)r * GATES + n0 + 16] = acc1[p] + bgate[n0 + 16];
  }
}

// --- Kernel 2: sequential LSTM scan, single workgroup, h in LDS ------------
// 1024 threads = 32 waves; wave w owns h columns [16w,16w+16) => 4 gate tiles.
__global__ void __launch_bounds__(1024)
lstm_scan(const float* __restrict__ xw,
          const bf16* __restrict__ Wh_t,           // [GATES][HID]
          const float* __restrict__ h_init,
          bf16* __restrict__ hs_bf) {
  __shared__ bf16 h_lds[NBATCH * HID];
  const int tid = threadIdx.x;
  const int lane = tid & 31;
  const int wave = tid >> 5;                       // 0..31
  for (int i = tid; i < NBATCH * HID; i += 1024)
    h_lds[i] = (bf16)h_init[i & (HID - 1)];
  __syncthreads();

  const int ncol = (wave << 4) + (lane & 15);      // h column 0..511
  const bf16* arow = h_lds + (lane & 15) * HID;
  float c[8];
#pragma unroll
  for (int p = 0; p < 8; ++p) c[p] = 0.f;

  for (int t = 0; t < T_STEPS; ++t) {
    v8f acc[4] = {{}, {}, {}, {}};
    for (int kk = 0; kk < HID / 32; ++kk) {
      v16bf a = loadA_bf(arow, kk, lane);          // ds_load_b128 x2
#pragma unroll
      for (int g = 0; g < 4; ++g) {
        v16bf bm = loadB_bf(Wh_t, g * HID + ncol, HID, kk, lane);
        acc[g] = wmma_bf16(a, bm, acc[g]);
      }
    }
    __syncthreads();   // all h_lds reads for step t complete
#pragma unroll
    for (int p = 0; p < 8; ++p) {
      const int mr = c_row(lane, p);
      const long r = (long)(t * 16 + mr);
      float ai = acc[0][p] + xw[r * GATES + 0 * HID + ncol];
      float af = acc[1][p] + xw[r * GATES + 1 * HID + ncol];
      float ao = acc[2][p] + xw[r * GATES + 2 * HID + ncol];
      float ag = acc[3][p] + xw[r * GATES + 3 * HID + ncol];
      float ig = 1.f / (1.f + __expf(-ai));
      float fg = 1.f / (1.f + __expf(-af));
      float og = 1.f / (1.f + __expf(-ao));
      float gg = tanhf(ag);
      float cn = fg * c[p] + ig * gg;
      c[p] = cn;
      bf16 hb = (bf16)(og * tanhf(cn));
      h_lds[mr * HID + ncol] = hb;
      hs_bf[r * HID + ncol]  = hb;
    }
    __syncthreads();   // new h visible before step t+1 reads
  }
}

// --- Kernel 3: vocab GEMM + fused chunk-local logsumexp / target capture ---
// grid (125 chunks, 256 row tiles), block 256 (8 waves, 2 col tiles/wave)
__global__ void vocab_logits(const bf16* __restrict__ hs_bf,     // [NROWS][HID]
                             const bf16* __restrict__ Wv_t,      // [VOCAB][HID]
                             const float* __restrict__ b_vocab,
                             const int* __restrict__ captions,
                             float* __restrict__ pmax,
                             float* __restrict__ psum,
                             float* __restrict__ ptgt) {
  __shared__ float tile[NBATCH][VCHUNK];
  __shared__ float redm[NBATCH][16];
  __shared__ float reds[NBATCH][16];
  __shared__ float redt[NBATCH];

  const int tid = threadIdx.x;
  const int lane = tid & 31;
  const int wave = tid >> 5;
  const int rt = blockIdx.y;
  const int chunk = blockIdx.x;

  if (tid < NBATCH) redt[tid] = 0.f;

  const bf16* arow = hs_bf + (long)(rt * 16 + (lane & 15)) * HID;
  const int nloc = wave * 32 + (lane & 15);
  const int gcol = chunk * VCHUNK + nloc;
  const bf16* brow0 = Wv_t + (long)gcol * HID;
  const bf16* brow1 = Wv_t + (long)(gcol + 16) * HID;

  v8f acc0 = {}, acc1 = {};
  for (int kk = 0; kk < HID / 32; ++kk) {
    __builtin_prefetch(brow0 + (kk << 5) + 128, 0, 1);  // stream next slab
    v16bf a = loadA_bf(arow, kk, lane);
    v16bf b0 = loadB_bf(Wv_t, gcol, HID, kk, lane);
    v16bf b1 = loadB_bf(Wv_t, gcol + 16, HID, kk, lane);
    acc0 = wmma_bf16(a, b0, acc0);
    acc1 = wmma_bf16(a, b1, acc1);
  }
  (void)brow1;
#pragma unroll
  for (int p = 0; p < 8; ++p) {
    int m = c_row(lane, p);
    tile[m][nloc]      = acc0[p] + b_vocab[gcol];
    tile[m][nloc + 16] = acc1[p] + b_vocab[gcol + 16];
  }
  __syncthreads();

  // 16 threads per row, each scans 16 columns
  const int rowm = tid & 15, seg = tid >> 4;
  const int tgt = captions[rowm * TP1 + rt + 1];   // cap_out for (n=rowm,t=rt)
  float mx = -3.4e38f;
#pragma unroll
  for (int j = 0; j < 16; ++j) mx = fmaxf(mx, tile[rowm][seg * 16 + j]);
  float sm = 0.f;
#pragma unroll
  for (int j = 0; j < 16; ++j) {
    int lc = seg * 16 + j;
    float v = tile[rowm][lc];
    sm += __expf(v - mx);
    if (chunk * VCHUNK + lc == tgt) redt[rowm] = v;  // unique writer per row
  }
  redm[rowm][seg] = mx;
  reds[rowm][seg] = sm;
  __syncthreads();

  if (tid < NBATCH) {
    float M = -3.4e38f;
#pragma unroll
    for (int s = 0; s < 16; ++s) M = fmaxf(M, redm[tid][s]);
    float S = 0.f;
#pragma unroll
    for (int s = 0; s < 16; ++s) S += __expf(redm[tid][s] - M) * reds[tid][s];
    long idx = (long)(rt * 16 + tid) * NCHUNKS + chunk;
    pmax[idx] = M;
    psum[idx] = S;
    ptgt[idx] = redt[tid];
  }
}

// --- Kernel 4: merge chunk partials -> masked NLL sum / N ------------------
__global__ void finalize_loss(const float* __restrict__ pmax,
                              const float* __restrict__ psum,
                              const float* __restrict__ ptgt,
                              const int* __restrict__ captions,
                              float* __restrict__ out) {
  __shared__ float red[256];
  const int tid = threadIdx.x;
  float local = 0.f;
  for (int r = tid; r < NROWS; r += 256) {
    int n = r & 15, t = r >> 4;
    int tgt = captions[n * TP1 + t + 1];
    if (tgt != 0) {                                  // NULL_TOKEN mask
      const float* pm = pmax + (long)r * NCHUNKS;
      const float* ps = psum + (long)r * NCHUNKS;
      const float* pt = ptgt + (long)r * NCHUNKS;
      float M = -3.4e38f;
      for (int c2 = 0; c2 < NCHUNKS; ++c2) M = fmaxf(M, pm[c2]);
      float S = 0.f, lt = 0.f;
      for (int c2 = 0; c2 < NCHUNKS; ++c2) {
        S += __expf(pm[c2] - M) * ps[c2];
        lt += pt[c2];                                // exactly one chunk != 0
      }
      local += logf(S) + M - lt;                     // -log p(target)
    }
  }
  red[tid] = local;
  __syncthreads();
  for (int off = 128; off > 0; off >>= 1) {
    if (tid < off) red[tid] += red[tid + off];
    __syncthreads();
  }
  if (tid == 0) out[0] = red[0] / (float)NBATCH;
}

// ---------------------------------------------------------------------------
extern "C" void kernel_launch(void* const* d_in, const int* in_sizes, int n_in,
                              void* d_out, int out_size, void* d_ws, size_t ws_size,
                              hipStream_t stream) {
  const int*   captions = (const int*)d_in[0];
  const float* W_embed  = (const float*)d_in[1];
  const float* Wx       = (const float*)d_in[2];   // [DWVEC][GATES]
  const float* Wh       = (const float*)d_in[3];   // [HID][GATES]
  const float* bgate    = (const float*)d_in[4];
  const float* W_vocab  = (const float*)d_in[5];   // [HID][VOCABSZ]
  const float* b_vocab  = (const float*)d_in[6];
  const float* h_init   = (const float*)d_in[7];
  float* out = (float*)d_out;

  // workspace carve-out (all offsets 256B aligned), total ~80 MB
  char* p = (char*)d_ws;
  bf16*  Wv_t   = (bf16*)p;  p += (size_t)HID * VOCABSZ * 2;   // 32.77 MB
  bf16*  Wh_t   = (bf16*)p;  p += (size_t)HID * GATES * 2;     //  2.10 MB
  bf16*  Wx_t   = (bf16*)p;  p += (size_t)DWVEC * GATES * 2;   //  1.05 MB
  float* xw     = (float*)p; p += (size_t)NROWS * GATES * 4;   // 33.55 MB
  bf16*  hs_bf  = (bf16*)p;  p += (size_t)NROWS * HID * 2;     //  4.19 MB
  float* pmax   = (float*)p; p += (size_t)NROWS * NCHUNKS * 4; //  2.05 MB
  float* psum   = (float*)p; p += (size_t)NROWS * NCHUNKS * 4;
  float* ptgt   = (float*)p; p += (size_t)NROWS * NCHUNKS * 4;

  cvt_transpose_bf16<<<dim3(HID / 32, VOCABSZ / 32), 256, 0, stream>>>(
      W_vocab, Wv_t, HID, VOCABSZ);
  cvt_transpose_bf16<<<dim3(HID / 32, GATES / 32), 256, 0, stream>>>(
      Wh, Wh_t, HID, GATES);
  cvt_transpose_bf16<<<dim3(DWVEC / 32, GATES / 32), 256, 0, stream>>>(
      Wx, Wx_t, DWVEC, GATES);

  embed_xw<<<dim3(GATES / 256, T_STEPS), 256, 0, stream>>>(
      captions, W_embed, Wx_t, bgate, xw);

  lstm_scan<<<1, 1024, 0, stream>>>(xw, Wh_t, h_init, hs_bf);

  vocab_logits<<<dim3(NCHUNKS, T_STEPS), 256, 0, stream>>>(
      hs_bf, Wv_t, b_vocab, captions, pmax, psum, ptgt);

  finalize_loss<<<1, 256, 0, stream>>>(pmax, psum, ptgt, captions, out);
}
